// AdaConv_23914377904427
// MI455X (gfx1250) — compile-verified, matrix operand order checked by reading, not run on hardware
//
#include <hip/hip_runtime.h>

typedef __bf16 bf16;
typedef __attribute__((ext_vector_type(8)))  __bf16 bf16x8;
typedef __attribute__((ext_vector_type(16))) __bf16 v16bf;
typedef __attribute__((ext_vector_type(8)))  float  v8f;

#define BATCH 8
#define HW    64
#define MPB   (HW*HW)   // 4096 spatial positions per sample
#define CH    512

__device__ __forceinline__ int refl(int i) {
    return i < 0 ? -i : (i > (HW-1) ? 2*(HW-1) - i : i);
}

// ---------------------------------------------------------------------------
// Instance norm: split-spatial partial sums with float atomics
// ---------------------------------------------------------------------------
__global__ void zero_f32_kernel(float* p, int n) {
    int i = blockIdx.x * 256 + threadIdx.x;
    if (i < n) p[i] = 0.f;
}

__global__ __launch_bounds__(256)
void in_stats_kernel(const float* __restrict__ x, float* __restrict__ sums) {
    // grid: (spatial chunks=16, channel chunks=2, B=8); block: 256 threads
    const int b = blockIdx.z;
    const int c = blockIdx.y * 256 + threadIdx.x;
    const long base = (((long)b * MPB) + blockIdx.x * 256) * CH + c;
    float s = 0.f, s2 = 0.f;
    for (int i = 0; i < 256; ++i) {
        float v = x[base + (long)i * CH];
        s += v; s2 += v * v;
    }
    atomicAdd(&sums[b * CH + c], s);
    atomicAdd(&sums[BATCH * CH + b * CH + c], s2);
}

__global__ void in_finalize_kernel(const float* __restrict__ sums,
                                   float* __restrict__ meanrstd) {
    int i = blockIdx.x * 256 + threadIdx.x;
    if (i >= BATCH * CH) return;
    float m = sums[i] * (1.f / MPB);
    float v = sums[BATCH * CH + i] * (1.f / MPB) - m * m;
    meanrstd[i] = m;
    meanrstd[BATCH * CH + i] = rsqrtf(v + 1e-5f);
}

__global__ __launch_bounds__(256)
void norm_cast_kernel(const float* __restrict__ x,
                      const float* __restrict__ meanrstd,
                      bf16* __restrict__ xn) {
    const long n = (long)BATCH * MPB * CH;
    for (long i = (long)blockIdx.x * 256 + threadIdx.x; i < n;
         i += (long)gridDim.x * 256) {
        int c = (int)(i & (CH - 1));          // CH = 512 (pow2)
        int b = (int)(i >> 21);               // MPB*CH = 2^21
        int s = b * CH + c;
        xn[i] = (bf16)((x[i] - meanrstd[s]) * meanrstd[BATCH * CH + s]);
    }
}

// ---------------------------------------------------------------------------
// Fused cast f32->bf16 + transpose of one 512x512 block:
//   src[ci][co] (f32, co contiguous)  ->  dst[co][ci] (bf16, ci contiguous)
// grid: (16, 16, nMatrices); block: 256
// ---------------------------------------------------------------------------
__global__ __launch_bounds__(256)
void cast_transpose_kernel(const float* __restrict__ src, bf16* __restrict__ dst) {
    __shared__ float tile[32][33];
    const long mat = blockIdx.z;
    const float* s = src + mat * (long)CH * CH;
    bf16*        d = dst + mat * (long)CH * CH;
    const int ci0 = blockIdx.x * 32;
    const int co0 = blockIdx.y * 32;
    const int tx = threadIdx.x & 31;
    const int ty = threadIdx.x >> 5;     // 0..7
#pragma unroll
    for (int i = 0; i < 4; ++i)
        tile[ty + 8 * i][tx] = s[(long)(ci0 + ty + 8 * i) * CH + co0 + tx];
    __syncthreads();
#pragma unroll
    for (int i = 0; i < 4; ++i)
        d[(long)(co0 + ty + 8 * i) * CH + ci0 + tx] = (bf16)tile[tx][ty + 8 * i];
}

// ---------------------------------------------------------------------------
// Implicit-GEMM conv (reflect pad) / pointwise GEMM via WMMA bf16
//   out[b, m, co] = sum_{tap, ci} in[b, reflect(m,tap), ci] * wT[b?, tap, co, ci]
//   M = 4096 (per sample), N = 512, K = TAPS*512
//   block: 256 threads = 8 waves; tile 128(M) x 256(N); K-step 32
//   wave grid 2(M) x 4(N): each wave 64x64 = 4x4 tiles of 16x16
//   Double-buffered LDS, one barrier per K-step.
// ---------------------------------------------------------------------------
#define LDSTRIDE 40                    // padded row stride (elements)
#define ABUFE    (128 * LDSTRIDE)      // elements per A LDS buffer
#define BBUFE    (256 * LDSTRIDE)      // elements per B LDS buffer

template<int TAPS>
__device__ __forceinline__ void load_step(int step, int t, int mtile, int ntile,
                                          const bf16* __restrict__ inB,
                                          const bf16* __restrict__ wB,
                                          bf16x8* Areg, bf16x8* Breg) {
    const int tap = (TAPS == 9) ? (step >> 4) : 0;
    const int kc  = (step & 15) * 32;
    const int dh  = (TAPS == 9) ? (tap / 3 - 1) : 0;
    const int dv  = (TAPS == 9) ? (tap % 3 - 1) : 0;
    // A: 128 rows x 32 ci  -> 512 chunks of 8 -> 2 per thread
#pragma unroll
    for (int q = 0; q < 2; ++q) {
        int linear = t + q * 256;
        int row = linear >> 2;           // 0..127
        int cq  = (linear & 3) * 8;
        int m = mtile * 128 + row;
        int hp = refl((m >> 6) + dh);
        int wp = refl((m & 63) + dv);
        Areg[q] = *(const bf16x8*)(inB + ((long)(hp * HW + wp)) * CH + kc + cq);
    }
    // B: 256 co rows x 32 ci -> 1024 chunks of 8 -> 4 per thread
#pragma unroll
    for (int q = 0; q < 4; ++q) {
        int linear = t + q * 256;
        int row = linear >> 2;           // 0..255 (co)
        int cq  = (linear & 3) * 8;
        Breg[q] = *(const bf16x8*)(wB +
                   ((long)(tap * CH + ntile * 256 + row)) * CH + kc + cq);
    }
}

__device__ __forceinline__ void store_step(int t, bf16* Ab, bf16* Bb,
                                           const bf16x8* Areg, const bf16x8* Breg) {
#pragma unroll
    for (int q = 0; q < 2; ++q) {
        int linear = t + q * 256;
        int row = linear >> 2;
        int cq  = (linear & 3) * 8;
        *(bf16x8*)(&Ab[row * LDSTRIDE + cq]) = Areg[q];
    }
#pragma unroll
    for (int q = 0; q < 4; ++q) {
        int linear = t + q * 256;
        int row = linear >> 2;
        int cq  = (linear & 3) * 8;
        *(bf16x8*)(&Bb[row * LDSTRIDE + cq]) = Breg[q];
    }
}

template<int TAPS, typename OutT>
__global__ __launch_bounds__(256)
void conv_gemm_kernel(const bf16* __restrict__ in,
                      const bf16* __restrict__ wgt,   // pre-transposed [tap][co][ci]
                      long wstride_b,
                      const float* __restrict__ bias,
                      int bias_stride_b,
                      OutT* __restrict__ out) {
    __shared__ bf16 Alds[2 * ABUFE];   // [buf][row(M)][k]   20 KB
    __shared__ bf16 Blds[2 * BBUFE];   // [buf][co(N)][k]    40 KB

    const int b     = blockIdx.z;
    const int mtile = blockIdx.y;     // 0..31
    const int ntile = blockIdx.x;     // 0..1
    const int t     = threadIdx.x;
    const int lane  = t & 31;
    const int wave  = t >> 5;         // 0..7
    const int wm    = wave >> 2;      // 0..1  (M)
    const int wn    = wave & 3;       // 0..3  (N)
    const int lhalf = lane >> 4;      // 0 | 1
    const int l16   = lane & 15;
    const int koff  = lhalf * 8;      // per-lane K base (ISA 16-bit layout)

    const bf16* __restrict__ inB = in  + (long)b * MPB * CH;
    const bf16* __restrict__ wB  = wgt + (long)b * wstride_b;

    v8f acc[16];
#pragma unroll
    for (int i = 0; i < 16; ++i) acc[i] = (v8f){0.f,0.f,0.f,0.f,0.f,0.f,0.f,0.f};

    const int NSTEPS = TAPS * 16;
    bf16x8 Areg[2], Breg[4];

    // prologue: stage step 0 and commit to buffer 0
    load_step<TAPS>(0, t, mtile, ntile, inB, wB, Areg, Breg);
    store_step(t, &Alds[0], &Blds[0], Areg, Breg);
    __syncthreads();

    for (int s = 0; s < NSTEPS; ++s) {
        const int cur = s & 1;
        const int nxt = cur ^ 1;
        if (s + 1 < NSTEPS)   // issue next tile's global loads early
            load_step<TAPS>(s + 1, t, mtile, ntile, inB, wB, Areg, Breg);

        // ---- fragments + WMMA from current buffer ----
        const bf16* Ab = &Alds[cur * ABUFE];
        const bf16* Bb = &Blds[cur * BBUFE];
        v16bf bfr[4];
#pragma unroll
        for (int nt = 0; nt < 4; ++nt) {
            int co = wn * 64 + nt * 16 + l16;
            bf16x8 lo = *(const bf16x8*)(&Bb[co * LDSTRIDE + koff]);
            bf16x8 hi = *(const bf16x8*)(&Bb[co * LDSTRIDE + koff + 16]);
            bfr[nt] = __builtin_shufflevector(lo, hi,
                0,1,2,3,4,5,6,7,8,9,10,11,12,13,14,15);
        }
#pragma unroll
        for (int mt = 0; mt < 4; ++mt) {
            int row = wm * 64 + mt * 16 + l16;
            bf16x8 lo = *(const bf16x8*)(&Ab[row * LDSTRIDE + koff]);
            bf16x8 hi = *(const bf16x8*)(&Ab[row * LDSTRIDE + koff + 16]);
            v16bf af = __builtin_shufflevector(lo, hi,
                0,1,2,3,4,5,6,7,8,9,10,11,12,13,14,15);
#pragma unroll
            for (int nt = 0; nt < 4; ++nt) {
                acc[mt * 4 + nt] = __builtin_amdgcn_wmma_f32_16x16x32_bf16(
                    false, af, false, bfr[nt],
                    (short)0, acc[mt * 4 + nt], false, false);
            }
        }

        if (s + 1 < NSTEPS)   // commit staged tile to the other buffer
            store_step(t, &Alds[nxt * ABUFE], &Blds[nxt * BBUFE], Areg, Breg);
        __syncthreads();
    }

    // ---- writeback: C layout — VGPR r, lanes 0-15: M=r, lanes 16-31: M=r+8
    const float* biasB = bias ? bias + (long)b * bias_stride_b : nullptr;
#pragma unroll
    for (int mt = 0; mt < 4; ++mt) {
#pragma unroll
        for (int nt = 0; nt < 4; ++nt) {
            v8f c = acc[mt * 4 + nt];
            int co = ntile * 256 + wn * 64 + nt * 16 + l16;
            float bv = biasB ? biasB[co] : 0.f;
#pragma unroll
            for (int r = 0; r < 8; ++r) {
                int m = mtile * 128 + wm * 64 + mt * 16 + r + lhalf * 8;
                out[((long)b * MPB + m) * CH + co] = (OutT)(c[r] + bv);
            }
        }
    }
}

// ---------------------------------------------------------------------------
extern "C" void kernel_launch(void* const* d_in, const int* in_sizes, int n_in,
                              void* d_out, int out_size, void* d_ws, size_t ws_size,
                              hipStream_t stream) {
    const float* x    = (const float*)d_in[0];   // [8,64,64,512]
    const float* dwk  = (const float*)d_in[1];   // [8,3,3,512,512]
    const float* pwk  = (const float*)d_in[2];   // [8,1,1,512,512]
    const float* bias = (const float*)d_in[3];   // [8,512]
    const float* cw   = (const float*)d_in[4];   // [3,3,512,512]
    const float* cb   = (const float*)d_in[5];   // [512]
    float* out = (float*)d_out;

    char*  ws  = (char*)d_ws;
    size_t off = 0;
    auto carve = [&](size_t bytes) -> void* {
        void* p = ws + off;
        off += (bytes + 255) & ~(size_t)255;
        return p;
    };
    const long NACT = (long)BATCH * MPB * CH;       // 16.78M
    float* sums     = (float*)carve(2 * BATCH * CH * sizeof(float));
    float* meanrstd = (float*)carve(2 * BATCH * CH * sizeof(float));
    bf16*  xn  = (bf16*)carve(NACT * sizeof(bf16));
    bf16*  y1  = (bf16*)carve(NACT * sizeof(bf16));
    bf16*  y2  = (bf16*)carve(NACT * sizeof(bf16));
    bf16*  dwb = (bf16*)carve((size_t)BATCH * 9 * CH * CH * sizeof(bf16));
    bf16*  pwb = (bf16*)carve((size_t)BATCH * CH * CH * sizeof(bf16));
    bf16*  cwb = (bf16*)carve((size_t)9 * CH * CH * sizeof(bf16));

    // ---- instance norm ----
    zero_f32_kernel<<<(2 * BATCH * CH + 255) / 256, 256, 0, stream>>>(
        sums, 2 * BATCH * CH);
    in_stats_kernel<<<dim3(16, 2, BATCH), 256, 0, stream>>>(x, sums);
    in_finalize_kernel<<<(BATCH * CH + 255) / 256, 256, 0, stream>>>(sums, meanrstd);
    norm_cast_kernel<<<4096, 256, 0, stream>>>(x, meanrstd, xn);

    // ---- weight casts f32 -> bf16, transposed to [..][co][ci] ----
    cast_transpose_kernel<<<dim3(16, 16, BATCH * 9), 256, 0, stream>>>(dwk, dwb);
    cast_transpose_kernel<<<dim3(16, 16, BATCH),     256, 0, stream>>>(pwk, pwb);
    cast_transpose_kernel<<<dim3(16, 16, 9),         256, 0, stream>>>(cw,  cwb);

    // ---- three GEMM stages ----
    dim3 grid(CH / 256, MPB / 128, BATCH);   // (2, 32, 8)
    // per-sample 3x3 conv, no bias -> y1 (bf16)
    conv_gemm_kernel<9, bf16><<<grid, 256, 0, stream>>>(
        xn, dwb, (long)9 * CH * CH, nullptr, 0, y1);
    // per-sample pointwise + per-sample bias -> y2 (bf16)
    conv_gemm_kernel<1, bf16><<<grid, 256, 0, stream>>>(
        y1, pwb, (long)CH * CH, bias, CH, y2);
    // shared 3x3 conv + bias -> out (f32)
    conv_gemm_kernel<9, float><<<grid, 256, 0, stream>>>(
        y2, cwb, 0, cb, 0, out);

    (void)in_sizes; (void)n_in; (void)out_size; (void)ws_size;
}
